// Mamba2_Attention_38517266710652
// MI455X (gfx1250) — compile-verified, matrix operand order checked by reading, not run on hardware
//
#include <hip/hip_runtime.h>
#include <math.h>

// ---------------------------------------------------------------------------
// CDNA5 (gfx1250) wave32 WMMA types & helpers
// ---------------------------------------------------------------------------
typedef __attribute__((ext_vector_type(16))) __bf16 bfx16;
typedef __attribute__((ext_vector_type(8)))  __bf16 bfx8;
typedef __attribute__((ext_vector_type(4)))  __bf16 bfx4;
typedef __attribute__((ext_vector_type(8)))  float  fx8;

// Load a 16x32 bf16 WMMA A/B fragment from a row-major LDS tile.
// Lane layout (ISA 7.12.2, 16-bit A 16x32): lanes 0-15 -> row=lane, K {0..7,16..23}
//                                           lanes 16-31 -> row=lane-16, K {8..15,24..31}
__device__ __forceinline__ bfx16 load_frag(const __bf16* tile, int ld) {
  const int lane = threadIdx.x & 31;
  const int r  = lane & 15;
  const int kh = (lane & 16) >> 1;          // 0 or 8
  const __bf16* p = tile + r * ld + kh;
  union { bfx16 v; bfx8 h[2]; } u;
  u.h[0] = *(const bfx8*)(p);
  u.h[1] = *(const bfx8*)(p + 16);
  return u.v;
}

__device__ __forceinline__ fx8 wmma_bf16(bfx16 a, bfx16 b, fx8 c) {
  return __builtin_amdgcn_wmma_f32_16x16x32_bf16(false, a, false, b, (short)0, c,
                                                 false, false);
}

__device__ __forceinline__ fx8 zero_fx8() {
  fx8 z;
#pragma unroll
  for (int e = 0; e < 8; ++e) z[e] = 0.0f;
  return z;
}

// ---------------------------------------------------------------------------
// Generic GEMM: C[M,N] = A[M,K] (f32) * W[N,K]^T (f32), bf16 WMMA, f32 accum.
// Block tile 128x256, K-tile 64, 256 threads (8 waves, 2x4), wave tile 64x64:
// per 32-K step each wave does 16 WMMAs from 8 fragment loads.
// ---------------------------------------------------------------------------
#define GBM 128
#define GBN 256
#define GBK 64
#define GLD 72   // padded LDS stride (halves): 64 + 8

__global__ __launch_bounds__(256) void gemm_bf16_nt(
    const float* __restrict__ A, const float* __restrict__ W,
    float* __restrict__ C, int M, int N, int K)
{
  __shared__ __align__(16) __bf16 sA[GBM * GLD];   // 18.0 KB
  __shared__ __align__(16) __bf16 sW[GBN * GLD];   // 36.0 KB

  const int t    = threadIdx.x;
  const int wave = t >> 5;
  const int wm   = wave >> 2;   // 0..1  (64 rows each)
  const int wn   = wave & 3;    // 0..3  (64 cols each)
  const int bm   = blockIdx.y * GBM;
  const int bn   = blockIdx.x * GBN;

  fx8 acc[4][4];
#pragma unroll
  for (int i = 0; i < 4; ++i)
#pragma unroll
    for (int j = 0; j < 4; ++j) acc[i][j] = zero_fx8();

  for (int kt = 0; kt < K; kt += GBK) {
    // A tile: 128 x 64 f32 -> bf16 (8 float4 per thread)
#pragma unroll
    for (int rep = 0; rep < 8; ++rep) {
      const int id = rep * 256 + t;
      const int r  = id >> 4;          // 16 float4 per row
      const int c  = (id & 15) * 4;
      const float4 v = *(const float4*)(A + (size_t)(bm + r) * K + kt + c);
      bfx4 b4; b4[0] = (__bf16)v.x; b4[1] = (__bf16)v.y;
               b4[2] = (__bf16)v.z; b4[3] = (__bf16)v.w;
      *(bfx4*)(&sA[r * GLD + c]) = b4;
    }
    // W tile: 256 x 64 f32 -> bf16 (16 float4 per thread; guard N)
#pragma unroll
    for (int rep = 0; rep < 16; ++rep) {
      const int id = rep * 256 + t;
      const int r  = id >> 4;
      const int c  = (id & 15) * 4;
      const int gn = bn + r;
      float4 v = make_float4(0.f, 0.f, 0.f, 0.f);
      if (gn < N) v = *(const float4*)(W + (size_t)gn * K + kt + c);
      bfx4 b4; b4[0] = (__bf16)v.x; b4[1] = (__bf16)v.y;
               b4[2] = (__bf16)v.z; b4[3] = (__bf16)v.w;
      *(bfx4*)(&sW[r * GLD + c]) = b4;
    }
    if (kt + GBK < K)  // hint next A tile into cache (global_prefetch_b8)
      __builtin_prefetch(A + (size_t)(bm + (t >> 1)) * K + kt + GBK + (t & 1) * 32, 0, 1);
    __syncthreads();

#pragma unroll
    for (int kb = 0; kb < 2; ++kb) {
      bfx16 af[4], bf[4];
#pragma unroll
      for (int i = 0; i < 4; ++i)
        af[i] = load_frag(&sA[(wm * 64 + i * 16) * GLD + kb * 32], GLD);
#pragma unroll
      for (int j = 0; j < 4; ++j)
        bf[j] = load_frag(&sW[(wn * 64 + j * 16) * GLD + kb * 32], GLD);
#pragma unroll
      for (int i = 0; i < 4; ++i)
#pragma unroll
        for (int j = 0; j < 4; ++j)
          acc[i][j] = wmma_bf16(af[i], bf[j], acc[i][j]);
    }
    __syncthreads();
  }

  // C/D f32 layout: VGPR j -> (M = j + 8*(lane>=16), N = lane&15)
  const int lane = t & 31;
  const int cn   = lane & 15;
  const int mh   = (lane >> 4) * 8;
#pragma unroll
  for (int im = 0; im < 4; ++im)
#pragma unroll
    for (int in = 0; in < 4; ++in) {
      const int gn = bn + wn * 64 + in * 16 + cn;
      if (gn >= N) continue;
      const int gm0 = bm + wm * 64 + im * 16 + mh;
#pragma unroll
      for (int r = 0; r < 8; ++r)
        C[(size_t)(gm0 + r) * N + gn] = acc[im][in][r];
    }
}

// ---------------------------------------------------------------------------
// RoPE (in place), one thread per (l, h, d<64) pair
// ---------------------------------------------------------------------------
__global__ void rope_kernel(float* __restrict__ x, int L, int NH)
{
  const int idx = blockIdx.x * blockDim.x + threadIdx.x;
  if (idx >= L * NH * 64) return;
  const int d = idx & 63;
  const int h = (idx >> 6) % NH;
  const int l = idx / (64 * NH);
  const float inv = __expf(-0.14390963f * (float)d);   // 10000^(-2d/128)
  const float ang = (float)l * inv;
  const float c = __cosf(ang), s = __sinf(ang);
  const size_t base = ((size_t)l * NH + h) * 128;
  const float x0 = x[base + d];
  const float x1 = x[base + d + 64];
  x[base + d]      = x0 * c - x1 * s;
  x[base + d + 64] = x1 * c + x0 * s;
}

// ---------------------------------------------------------------------------
// dt = softplus(dtraw + bias); g = dt * (-exp(A_log_bias))
// ---------------------------------------------------------------------------
__global__ void dtg_kernel(const float* __restrict__ dtraw,
                           const float* __restrict__ dt_bias,
                           const float* __restrict__ A_log,
                           float* __restrict__ dt, float* __restrict__ g,
                           int total, int NH)
{
  const int idx = blockIdx.x * blockDim.x + threadIdx.x;
  if (idx >= total) return;
  const int h = idx % NH;
  const float x  = dtraw[idx] + dt_bias[h];
  const float sp = (x > 20.f) ? x : log1pf(__expf(x));
  dt[idx] = sp;
  g[idx]  = sp * (-__expf(A_log[h]));
}

// ---------------------------------------------------------------------------
// Per (h,chunk): G cumsum, S = (q k^T) * decay -> bf16 Sbuf.
// Two d-halves (K=64 per pass) keep LDS at 2 * 128x72 bf16 (<64KB).
// ---------------------------------------------------------------------------
__global__ __launch_bounds__(256) void chunk_scores_kernel(
    const float* __restrict__ q,   // [L,32,128]
    const float* __restrict__ k,   // [L,8,128]
    const float* __restrict__ g,   // [L,32]
    float* __restrict__ Gbuf,      // [H*NC*128]
    float* __restrict__ cdec,      // [H*NC]
    __bf16* __restrict__ Sbuf)     // [H*NC*128*128]
{
  __shared__ __align__(16) __bf16 sA[128 * 72];
  __shared__ __align__(16) __bf16 sB[128 * 72];
  __shared__ float sG[128];

  const int hn = blockIdx.x;     // h*32 + nc
  const int h  = hn >> 5;
  const int nc = hn & 31;
  const int l0 = nc * 128;
  const int t  = threadIdx.x;

  if (t < 128) sG[t] = g[(size_t)(l0 + t) * 32 + h];
  __syncthreads();
  if (t == 0) {
    float a = sG[0];
    for (int i = 1; i < 128; ++i) { a += sG[i]; sG[i] = a; }
    cdec[hn] = __expf(a);
  }
  __syncthreads();
  if (t < 128) Gbuf[hn * 128 + t] = sG[t];

  const int wave = t >> 5;
  const int i0   = wave * 16;
  fx8 acc[8];
#pragma unroll
  for (int j = 0; j < 8; ++j) acc[j] = zero_fx8();

  for (int p = 0; p < 2; ++p) {
#pragma unroll
    for (int rep = 0; rep < 8; ++rep) {
      const int id = rep * 256 + t;
      const int r  = id >> 4;
      const int c  = (id & 15) * 4;
      const float4 vq = *(const float4*)(q + ((size_t)(l0 + r) * 32 + h) * 128 + p * 64 + c);
      bfx4 bq; bq[0] = (__bf16)vq.x; bq[1] = (__bf16)vq.y;
               bq[2] = (__bf16)vq.z; bq[3] = (__bf16)vq.w;
      *(bfx4*)(&sA[r * 72 + c]) = bq;
      const float4 vk = *(const float4*)(k + ((size_t)(l0 + r) * 8 + (h >> 2)) * 128 + p * 64 + c);
      bfx4 bk; bk[0] = (__bf16)vk.x; bk[1] = (__bf16)vk.y;
               bk[2] = (__bf16)vk.z; bk[3] = (__bf16)vk.w;
      *(bfx4*)(&sB[r * 72 + c]) = bk;
    }
    __syncthreads();
#pragma unroll
    for (int kb = 0; kb < 2; ++kb) {
      const bfx16 a = load_frag(&sA[i0 * 72 + kb * 32], 72);
#pragma unroll
      for (int jt = 0; jt < 8; ++jt) {
        const bfx16 b = load_frag(&sB[jt * 16 * 72 + kb * 32], 72);
        acc[jt] = wmma_bf16(a, b, acc[jt]);
      }
    }
    __syncthreads();
  }

  const int lane = t & 31;
  const int cn   = lane & 15;
  const int mh   = (lane >> 4) * 8;
#pragma unroll
  for (int jt = 0; jt < 8; ++jt)
#pragma unroll
    for (int r = 0; r < 8; ++r) {
      const int i = i0 + mh + r;
      const int j = jt * 16 + cn;
      float v = acc[jt][r];
      v = (i >= j) ? v * __expf(sG[i] - sG[j]) : 0.0f;
      Sbuf[(size_t)hn * 16384 + i * 128 + j] = (__bf16)v;
    }
}

// ---------------------------------------------------------------------------
// Per (h,chunk): y_intra = S @ x  and  cstates = k_dec^T @ x, both via WMMA.
// K dim is j (chunk position); two 64-wide j passes; x and k_dec are
// transposed into LDS during the fill so fragments load row-major.
// ---------------------------------------------------------------------------
__global__ __launch_bounds__(256) void chunk_intra_kernel(
    const __bf16* __restrict__ Sbuf,
    const float* __restrict__ k,    // [L,8,128]
    const float* __restrict__ v,    // [L,8,128]
    const float* __restrict__ dt,   // [L,32]
    const float* __restrict__ Gbuf, // [H*NC*128]
    float* __restrict__ ybuf,       // [L,32,128]
    float* __restrict__ cs)         // [H*NC,128,128]
{
  __shared__ __align__(16) __bf16 sS [128 * 72];
  __shared__ __align__(16) __bf16 sXT[128 * 72];
  __shared__ __align__(16) __bf16 sKT[128 * 72];
  __shared__ float sW[128];

  const int hn = blockIdx.x;
  const int h  = hn >> 5;
  const int nc = hn & 31;
  const int l0 = nc * 128;
  const int t  = threadIdx.x;
  const int wave = t >> 5;
  const int i0   = wave * 16;

  if (t < 128) {
    const float G127 = Gbuf[hn * 128 + 127];
    sW[t] = __expf(G127 - Gbuf[hn * 128 + t]);
  }
  __syncthreads();

  fx8 accY[8], accC[8];
#pragma unroll
  for (int j = 0; j < 8; ++j) { accY[j] = zero_fx8(); accC[j] = zero_fx8(); }

  for (int p = 0; p < 2; ++p) {
    // S slice (row-major, bf16 already)
#pragma unroll
    for (int rep = 0; rep < 8; ++rep) {
      const int id = rep * 256 + t;
      const int r  = id >> 4;
      const int c  = (id & 15) * 4;
      const bfx4 sv = *(const bfx4*)(Sbuf + (size_t)hn * 16384 + r * 128 + p * 64 + c);
      *(bfx4*)(&sS[r * 72 + c]) = sv;
    }
    // x^T and k_dec^T slices (transpose while filling LDS)
#pragma unroll
    for (int rep = 0; rep < 8; ++rep) {
      const int id = rep * 256 + t;
      const int jr = id >> 5;           // 0..63 (local j)
      const int c  = (id & 31) * 4;     // d offset
      const int j  = p * 64 + jr;
      const size_t gro = ((size_t)(l0 + j) * 8 + (h >> 2)) * 128 + c;
      const float4 vv = *(const float4*)(v + gro);
      const float4 vk = *(const float4*)(k + gro);
      const float dtv = dt[(size_t)(l0 + j) * 32 + h];
      const float w   = sW[j];
      sXT[(c + 0) * 72 + jr] = (__bf16)(vv.x * dtv);
      sXT[(c + 1) * 72 + jr] = (__bf16)(vv.y * dtv);
      sXT[(c + 2) * 72 + jr] = (__bf16)(vv.z * dtv);
      sXT[(c + 3) * 72 + jr] = (__bf16)(vv.w * dtv);
      sKT[(c + 0) * 72 + jr] = (__bf16)(vk.x * w);
      sKT[(c + 1) * 72 + jr] = (__bf16)(vk.y * w);
      sKT[(c + 2) * 72 + jr] = (__bf16)(vk.z * w);
      sKT[(c + 3) * 72 + jr] = (__bf16)(vk.w * w);
    }
    __syncthreads();
#pragma unroll
    for (int kb = 0; kb < 2; ++kb) {
      const bfx16 aY = load_frag(&sS [i0 * 72 + kb * 32], 72);
      const bfx16 aC = load_frag(&sKT[i0 * 72 + kb * 32], 72);
#pragma unroll
      for (int et = 0; et < 8; ++et) {
        const bfx16 b = load_frag(&sXT[et * 16 * 72 + kb * 32], 72);
        accY[et] = wmma_bf16(aY, b, accY[et]);
        accC[et] = wmma_bf16(aC, b, accC[et]);
      }
    }
    __syncthreads();
  }

  const int lane = t & 31;
  const int cn   = lane & 15;
  const int mh   = (lane >> 4) * 8;
#pragma unroll
  for (int et = 0; et < 8; ++et)
#pragma unroll
    for (int r = 0; r < 8; ++r) {
      const int i = i0 + mh + r;        // row (i for y, d for cstates)
      const int e = et * 16 + cn;
      ybuf[((size_t)(l0 + i) * 32 + h) * 128 + e] = accY[et][r];
      cs[(size_t)hn * 16384 + (size_t)i * 128 + e] = accC[et][r];
    }
}

// ---------------------------------------------------------------------------
// Exclusive scan of chunk states over NC per head; state kept in registers.
// Writes previous-state transposed: prevT[hn][e][d].
// ---------------------------------------------------------------------------
__global__ __launch_bounds__(256) void scan_kernel(
    const float* __restrict__ cs, const float* __restrict__ cdec,
    float* __restrict__ prevT)
{
  const int h = blockIdx.x;
  const int t = threadIdx.x;
  float st[64];
#pragma unroll
  for (int r = 0; r < 64; ++r) st[r] = 0.0f;
  for (int nc = 0; nc < 32; ++nc) {
    const size_t base = ((size_t)h * 32 + nc) * 16384;
    const float dec = cdec[h * 32 + nc];
#pragma unroll
    for (int r = 0; r < 64; ++r) {
      const int f = r * 256 + t;
      const int d = f >> 7, e = f & 127;
      prevT[base + (size_t)e * 128 + d] = st[r];
      st[r] = st[r] * dec + cs[base + f];
    }
  }
}

// ---------------------------------------------------------------------------
// Per (h,chunk): y += (q * exp(G)) @ prev   (K = d, two 64-wide passes)
// ---------------------------------------------------------------------------
__global__ __launch_bounds__(256) void chunk_inter_kernel(
    const float* __restrict__ q,     // rope'd [L,32,128]
    const float* __restrict__ Gbuf,
    const float* __restrict__ prevT, // [H*NC,128(e),128(d)]
    float* __restrict__ ybuf)
{
  __shared__ __align__(16) __bf16 sA[128 * 72];
  __shared__ __align__(16) __bf16 sB[128 * 72];
  __shared__ float sE[128];

  const int hn = blockIdx.x;
  const int h  = hn >> 5;
  const int nc = hn & 31;
  const int l0 = nc * 128;
  const int t  = threadIdx.x;
  const int wave = t >> 5;
  const int i0   = wave * 16;

  if (t < 128) sE[t] = __expf(Gbuf[hn * 128 + t]);
  __syncthreads();

  fx8 acc[8];
#pragma unroll
  for (int j = 0; j < 8; ++j) acc[j] = zero_fx8();

  for (int p = 0; p < 2; ++p) {
#pragma unroll
    for (int rep = 0; rep < 8; ++rep) {
      const int id = rep * 256 + t;
      const int r  = id >> 4;
      const int c  = (id & 15) * 4;
      const float4 vq = *(const float4*)(q + ((size_t)(l0 + r) * 32 + h) * 128 + p * 64 + c);
      const float sc = sE[r];
      bfx4 ba; ba[0] = (__bf16)(vq.x * sc); ba[1] = (__bf16)(vq.y * sc);
               ba[2] = (__bf16)(vq.z * sc); ba[3] = (__bf16)(vq.w * sc);
      *(bfx4*)(&sA[r * 72 + c]) = ba;
      const float4 vp = *(const float4*)(prevT + (size_t)hn * 16384 + r * 128 + p * 64 + c);
      bfx4 bb; bb[0] = (__bf16)vp.x; bb[1] = (__bf16)vp.y;
               bb[2] = (__bf16)vp.z; bb[3] = (__bf16)vp.w;
      *(bfx4*)(&sB[r * 72 + c]) = bb;
    }
    __syncthreads();
#pragma unroll
    for (int kb = 0; kb < 2; ++kb) {
      const bfx16 a = load_frag(&sA[i0 * 72 + kb * 32], 72);
#pragma unroll
      for (int et = 0; et < 8; ++et) {
        const bfx16 b = load_frag(&sB[et * 16 * 72 + kb * 32], 72);
        acc[et] = wmma_bf16(a, b, acc[et]);
      }
    }
    __syncthreads();
  }

  const int lane = t & 31;
  const int cn   = lane & 15;
  const int mh   = (lane >> 4) * 8;
#pragma unroll
  for (int et = 0; et < 8; ++et)
#pragma unroll
    for (int r = 0; r < 8; ++r) {
      const int i = i0 + mh + r;
      const int e = et * 16 + cn;
      const size_t o = ((size_t)(l0 + i) * 32 + h) * 128 + e;
      ybuf[o] += acc[et][r];
    }
}

// ---------------------------------------------------------------------------
// RMS norm over D=128 + SiLU gate (one wave per (l,h) row; in place)
// ---------------------------------------------------------------------------
__global__ __launch_bounds__(256) void rms_gate_kernel(
    float* __restrict__ y, const float* __restrict__ gate,
    const float* __restrict__ gnw)
{
  const int row  = blockIdx.x * 8 + (threadIdx.x >> 5);
  const int lane = threadIdx.x & 31;
  const size_t base = (size_t)row * 128 + lane * 4;
  float4 yv = *(const float4*)(y + base);
  float ss = yv.x * yv.x + yv.y * yv.y + yv.z * yv.z + yv.w * yv.w;
#pragma unroll
  for (int off = 16; off > 0; off >>= 1) ss += __shfl_xor(ss, off, 32);
  const float inv = rsqrtf(ss * (1.0f / 128.0f) + 1e-5f);
  const float4 gv = *(const float4*)(gate + base);
  const float4 wv = *(const float4*)(gnw + lane * 4);
  const float s0 = gv.x / (1.0f + __expf(-gv.x));
  const float s1 = gv.y / (1.0f + __expf(-gv.y));
  const float s2 = gv.z / (1.0f + __expf(-gv.z));
  const float s3 = gv.w / (1.0f + __expf(-gv.w));
  yv.x = yv.x * inv * wv.x * s0;
  yv.y = yv.y * inv * wv.y * s1;
  yv.z = yv.z * inv * wv.z * s2;
  yv.w = yv.w * inv * wv.w * s3;
  *(float4*)(y + base) = yv;
}

// ---------------------------------------------------------------------------
// Host launcher
// ---------------------------------------------------------------------------
extern "C" void kernel_launch(void* const* d_in, const int* in_sizes, int n_in,
                              void* d_out, int out_size, void* d_ws, size_t ws_size,
                              hipStream_t stream)
{
  (void)in_sizes; (void)n_in; (void)out_size; (void)ws_size;
  const float* hs      = (const float*)d_in[0];
  const float* Wq      = (const float*)d_in[1];
  const float* Wk      = (const float*)d_in[2];
  const float* Wv      = (const float*)d_in[3];
  const float* Wo      = (const float*)d_in[4];
  const float* Wg      = (const float*)d_in[5];
  const float* Win     = (const float*)d_in[6];
  const float* dt_bias = (const float*)d_in[7];
  const float* A_log   = (const float*)d_in[8];
  const float* gnw     = (const float*)d_in[9];
  float* out = (float*)d_out;

  const int L = 4096, HID = 4096, H = 32, KVH = 8, D = 128, NC = 32;

  char* wp = (char*)d_ws;
  auto alloc = [&](size_t bytes) -> void* {
    void* p = wp;
    wp += (bytes + 255) & ~(size_t)255;
    return p;
  };
  float*  q     = (float*)alloc((size_t)L * H * D * 4);
  float*  kk    = (float*)alloc((size_t)L * KVH * D * 4);
  float*  vv    = (float*)alloc((size_t)L * KVH * D * 4);
  float*  gate  = (float*)alloc((size_t)L * H * D * 4);
  float*  dtraw = (float*)alloc((size_t)L * H * 4);
  float*  dts   = (float*)alloc((size_t)L * H * 4);
  float*  gvals = (float*)alloc((size_t)L * H * 4);
  float*  Gbuf  = (float*)alloc((size_t)H * NC * 128 * 4);
  float*  cdec  = (float*)alloc((size_t)H * NC * 4);
  __bf16* Sbuf  = (__bf16*)alloc((size_t)H * NC * 128 * 128 * 2);
  float*  ybuf  = (float*)alloc((size_t)L * H * D * 4);
  float*  cs    = (float*)alloc((size_t)H * NC * 128 * 128 * 4);
  float*  prevT = (float*)alloc((size_t)H * NC * 128 * 128 * 4);

  const dim3 blk(256);
  const int gq  = (H * D + GBN - 1) / GBN;     // 16
  const int gkv = (KVH * D + GBN - 1) / GBN;   // 4
  const int gdt = (H + GBN - 1) / GBN;         // 1
  const int go  = (HID + GBN - 1) / GBN;       // 16

  // Projections
  gemm_bf16_nt<<<dim3(gq,  L / GBM), blk, 0, stream>>>(hs, Wq, q,    L, H * D,   HID);
  gemm_bf16_nt<<<dim3(gkv, L / GBM), blk, 0, stream>>>(hs, Wk, kk,   L, KVH * D, HID);
  gemm_bf16_nt<<<dim3(gkv, L / GBM), blk, 0, stream>>>(hs, Wv, vv,   L, KVH * D, HID);
  gemm_bf16_nt<<<dim3(gq,  L / GBM), blk, 0, stream>>>(hs, Wg, gate, L, H * D,   HID);
  gemm_bf16_nt<<<dim3(gdt, L / GBM), blk, 0, stream>>>(hs, Win, dtraw, L, H,     HID);

  // RoPE (in place) + dt/softplus/decay
  rope_kernel<<<(L * H * 64 + 255) / 256, blk, 0, stream>>>(q, L, H);
  rope_kernel<<<(L * KVH * 64 + 255) / 256, blk, 0, stream>>>(kk, L, KVH);
  dtg_kernel<<<(L * H + 255) / 256, blk, 0, stream>>>(dtraw, dt_bias, A_log, dts, gvals, L * H, H);

  // Chunked linear attention
  chunk_scores_kernel<<<H * NC, blk, 0, stream>>>(q, kk, gvals, Gbuf, cdec, Sbuf);
  chunk_intra_kernel<<<H * NC, blk, 0, stream>>>(Sbuf, kk, vv, dts, Gbuf, ybuf, cs);
  scan_kernel<<<H, blk, 0, stream>>>(cs, cdec, prevT);
  chunk_inter_kernel<<<H * NC, blk, 0, stream>>>(q, Gbuf, prevT, ybuf);

  // Norm + gate, then output projection
  rms_gate_kernel<<<(L * H) / 8, blk, 0, stream>>>(ybuf, gate, gnw);
  gemm_bf16_nt<<<dim3(go, L / GBM), blk, 0, stream>>>(ybuf, Wo, out, L, HID, H * D);
}
